// MixtureOfExperts_61323543052832
// MI455X (gfx1250) — matrix operand order repaired and checked by reading, not compiled, not run on hardware
//
#include <hip/hip_runtime.h>
#include <hip/hip_bf16.h>
#include <math.h>

// ---------------------------------------------------------------------------
// MoE: B=4,S=1024,H=256,E=64,F=1024,TOP_K=2  -> T=4096 tokens
// Memory-bound on 128MB fp32 expert weights (~5.8us HBM floor @23.3TB/s).
// bf16 WMMA (v_wmma_f32_16x16x32_bf16) for both expert GEMMs; staging via
// b128 global loads, operand fragments via paired ds_load_b128.
// ---------------------------------------------------------------------------

#define T_TOK 4096
#define H_DIM 256
#define E_NUM 64
#define F_DIM 1024

typedef __attribute__((ext_vector_type(16))) __bf16 bf16x16;
typedef __attribute__((ext_vector_type(4)))  __bf16 bf16x4;
typedef __attribute__((ext_vector_type(8)))  float  f32x8;

// LDS strides (bf16 elements). All multiples of 8 -> 16B-aligned b128 frags.
#define XT_STR 264   // x tile   [32 tok][256 h]
#define W1_STR 264   // w1^T     [64 f ][256 h]
#define W2_STR 72    // w2^T     [256 h][64 f]
#define HD_STR 72    // hidden   [32 tok][64 f]
#define SMEM_META_OFF 92160                 // bytes: end of bf16 regions
#define SMEM_BYTES    (SMEM_META_OFF + 32*4 + 32*4 + 32*4)

static __device__ __forceinline__ f32x8 zero8() {
  f32x8 z = {0.f,0.f,0.f,0.f,0.f,0.f,0.f,0.f};
  return z;
}

// Load a 32-byte WMMA operand fragment as two contiguous 16B LDS reads.
static __device__ __forceinline__ bf16x16 ld_frag(const __bf16* p, int second_off_elems) {
  union { bf16x16 v; uint4 u[2]; } f;
  f.u[0] = *(const uint4*)(p);
  f.u[1] = *(const uint4*)(p + second_off_elems);
  return f.v;
}

static __device__ __forceinline__ f32x8 wmma_bf16(bf16x16 a, bf16x16 b, f32x8 c) {
  return __builtin_amdgcn_wmma_f32_16x16x32_bf16(false, a, false, b, (short)0, c, false, false);
}

// ------------------------------ router -------------------------------------
// One block (64 threads) per token: logits, softmax, top-2, renormalized gates.
__global__ void moe_router(const float* __restrict__ x, const float* __restrict__ wr,
                           const float* __restrict__ br, int* __restrict__ assign_e,
                           float* __restrict__ assign_g) {
  const int t = blockIdx.x;
  const int e = threadIdx.x;                  // 0..63
  __shared__ float xl[H_DIM];
  __shared__ float sl[E_NUM];
  for (int i = e; i < H_DIM; i += 64) xl[i] = x[t * H_DIM + i];
  __syncthreads();
  float acc = br[e];
  #pragma unroll 8
  for (int h = 0; h < H_DIM; ++h) acc += xl[h] * wr[h * E_NUM + e];
  sl[e] = acc;
  __syncthreads();
  if (e == 0) {
    float mx = sl[0];
    for (int i = 1; i < E_NUM; ++i) mx = fmaxf(mx, sl[i]);
    for (int i = 0; i < E_NUM; ++i) sl[i] = expf(sl[i] - mx);
    int i1 = 0; float v1 = sl[0];
    for (int i = 1; i < E_NUM; ++i) if (sl[i] > v1) { v1 = sl[i]; i1 = i; }
    int i2 = -1; float v2 = -1.f;
    for (int i = 0; i < E_NUM; ++i) if (i != i1 && sl[i] > v2) { v2 = sl[i]; i2 = i; }
    float inv = 1.f / (v1 + v2);              // softmax denom cancels in renorm
    assign_e[2 * t]     = i1;  assign_g[2 * t]     = v1 * inv;
    assign_e[2 * t + 1] = i2;  assign_g[2 * t + 1] = v2 * inv;
  }
}

// ------------------------- deterministic compaction ------------------------
// Block e scans all tokens in order; LDS prefix-sum -> stable bucket order.
__global__ void moe_compact(const int* __restrict__ assign_e, const float* __restrict__ assign_g,
                            int* __restrict__ btok, float* __restrict__ bgate,
                            int* __restrict__ counts) {
  const int e = blockIdx.x;
  const int tid = threadIdx.x;                // 256 threads
  __shared__ int sc[256];
  __shared__ int sbase;
  if (tid == 0) sbase = 0;
  __syncthreads();
  for (int c = 0; c < T_TOK / 256; ++c) {
    const int t = c * 256 + tid;
    const int e0 = assign_e[2 * t], e1 = assign_e[2 * t + 1];
    const int slot = (e0 == e) ? 0 : ((e1 == e) ? 1 : -1);  // top-2 distinct
    const int v = (slot >= 0) ? 1 : 0;
    sc[tid] = v;
    __syncthreads();
    for (int off = 1; off < 256; off <<= 1) {
      int tv = (tid >= off) ? sc[tid - off] : 0;
      __syncthreads();
      sc[tid] += tv;
      __syncthreads();
    }
    if (slot >= 0) {
      int pos = sbase + sc[tid] - 1;
      btok[e * T_TOK + pos]  = t | (slot << 16);
      bgate[e * T_TOK + pos] = assign_g[2 * t + slot];
    }
    __syncthreads();
    if (tid == 0) sbase += sc[255];
    __syncthreads();
  }
  if (tid == 0) counts[e] = sbase;
}

// ------------------------------ expert FFN ---------------------------------
// Block = (expert e, tile of 32 tokens), 256 threads = 8 wave32.
// Per F-chunk(64): stage bf16 w1^T/w2^T in LDS (b128 global loads),
// GEMM1+GELU -> LDS hid, GEMM2 accumulates [32 tok x 256 h] in VGPRs.
__global__ void __launch_bounds__(256)
moe_expert(const float* __restrict__ x,
           const float* __restrict__ w1g, const float* __restrict__ b1g,
           const float* __restrict__ w2g, const float* __restrict__ b2g,
           const int* __restrict__ btok, const float* __restrict__ bgate,
           const int* __restrict__ counts, float* __restrict__ contrib) {
  const int e    = blockIdx.x;
  const int tile = blockIdx.y;
  const int cnt  = counts[e];
  if (tile * 32 >= cnt) return;

  const int tid   = threadIdx.x;
  const int lane  = tid & 31;
  const int wid   = tid >> 5;
  const int lhalf = lane >> 4;                // 0 | 1
  const int l16   = lane & 15;

  extern __shared__ char smem[];
  __bf16* xt   = (__bf16*)smem;               // [32][XT_STR]
  __bf16* w1t  = xt  + 32 * XT_STR;           // [64][W1_STR]  (f-major: w1^T)
  __bf16* w2t  = w1t + 64 * W1_STR;           // [256][W2_STR] (h-major: w2^T)
  __bf16* hid  = w2t + 256 * W2_STR;          // [32][HD_STR]
  int*   stok  = (int*)(smem + SMEM_META_OFF);
  int*   sslot = stok + 32;
  float* sgate = (float*)(sslot + 32);

  // --- stage token metadata ---
  if (tid < 32) {
    int s = tile * 32 + tid;
    if (s < cnt) {
      int p = btok[e * T_TOK + s];
      stok[tid]  = p & 0xFFFF;
      sslot[tid] = (p >> 16) & 1;
      sgate[tid] = bgate[e * T_TOK + s];
    } else { stok[tid] = -1; sslot[tid] = 0; sgate[tid] = 0.f; }
  }
  // --- stage x tile (b128 fp32 loads -> packed bf16x4 = ds_store_b64) ---
  {
    const int c4 = (tid & 63) * 4;            // h column * 4
    const int rb = tid >> 6;                  // 0..3
    #pragma unroll
    for (int it = 0; it < 8; ++it) {
      const int i = it * 4 + rb;              // token row in tile
      const int s = tile * 32 + i;
      float4 v = make_float4(0.f, 0.f, 0.f, 0.f);
      if (s < cnt) {
        const int tok = btok[e * T_TOK + s] & 0xFFFF;
        v = *(const float4*)(x + (size_t)tok * H_DIM + c4);
      }
      bf16x4 p;
      p.x = (__bf16)v.x; p.y = (__bf16)v.y; p.z = (__bf16)v.z; p.w = (__bf16)v.w;
      *(bf16x4*)(xt + i * XT_STR + c4) = p;
    }
  }

  const float* w1e = w1g + (size_t)e * H_DIM * F_DIM;
  const float* w2e = w2g + (size_t)e * F_DIM * H_DIM;

  f32x8 c2[4];
  c2[0] = zero8(); c2[1] = zero8(); c2[2] = zero8(); c2[3] = zero8();

  for (int f0 = 0; f0 < F_DIM; f0 += 64) {
    __syncthreads();   // protects xt (1st iter) and w1t/w2t/hid from prior readers

    // stage w1 chunk [256 h][64 f] -> w1t[f][h]; b128 reads along f
    {
      const int f4 = (tid & 15) * 4;          // f * 4
      const int hb = tid >> 4;                // 0..15
      #pragma unroll 4
      for (int it = 0; it < 16; ++it) {
        const int h = it * 16 + hb;
        float4 v = *(const float4*)(w1e + (size_t)h * F_DIM + f0 + f4);
        w1t[(f4 + 0) * W1_STR + h] = (__bf16)v.x;
        w1t[(f4 + 1) * W1_STR + h] = (__bf16)v.y;
        w1t[(f4 + 2) * W1_STR + h] = (__bf16)v.z;
        w1t[(f4 + 3) * W1_STR + h] = (__bf16)v.w;
      }
    }
    // stage w2 chunk rows [f0..f0+64) of [1024 f][256 h] -> w2t[h][f]; b128 reads along h
    {
      const int h4 = (tid & 63) * 4;          // h * 4
      const int fb = tid >> 6;                // 0..3
      #pragma unroll 4
      for (int it = 0; it < 16; ++it) {
        const int f = it * 4 + fb;
        float4 v = *(const float4*)(w2e + (size_t)(f0 + f) * H_DIM + h4);
        w2t[(h4 + 0) * W2_STR + f] = (__bf16)v.x;
        w2t[(h4 + 1) * W2_STR + f] = (__bf16)v.y;
        w2t[(h4 + 2) * W2_STR + f] = (__bf16)v.z;
        w2t[(h4 + 3) * W2_STR + f] = (__bf16)v.w;
      }
    }
    if (f0 + 64 < F_DIM) {                    // global_prefetch_b8 next chunk
      __builtin_prefetch(w1e + f0 + 64, 0, 1);
      __builtin_prefetch(w2e + (size_t)(f0 + 64) * H_DIM, 0, 1);
    }
    __syncthreads();

    // GEMM1: hid[32 tok x 64 f] = x[32 x 256] * w1[256 x 64]; waves = 2 tok x 4 f-tiles
    {
      const int tokg  = wid & 1;
      const int ftile = wid >> 1;
      f32x8 c1 = zero8();
      #pragma unroll
      for (int k = 0; k < 8; ++k) {           // K over H=256, 32/step
        const __bf16* ap = xt  + (tokg * 16 + l16) * XT_STR + k * 32 + lhalf * 8;
        const __bf16* bp = w1t + (ftile * 16 + l16) * W1_STR + k * 32 + lhalf * 8;
        bf16x16 a = ld_frag(ap, 16);          // A: K {base, base+16}
        bf16x16 b = ld_frag(bp, 16);          // B: K {base, base+16}
        c1 = wmma_bf16(a, b, c1);
      }
      const float b1v = b1g[e * F_DIM + f0 + ftile * 16 + l16];
      #pragma unroll
      for (int r = 0; r < 8; ++r) {
        float v = c1[r] + b1v;
        float g = 0.5f * v * (1.f + erff(v * 0.70710678118654752f));  // exact GELU
        hid[(tokg * 16 + lhalf * 8 + r) * HD_STR + ftile * 16 + l16] = (__bf16)g;
      }
    }
    __syncthreads();

    // GEMM2: out += hid[32 x 64] * w2[64 x 256]; waves = 2 tok x 4 h-quarters
    {
      const int tokg = wid & 1;
      const int hq   = wid >> 1;
      #pragma unroll
      for (int ks = 0; ks < 2; ++ks) {        // K over chunk F=64, 32/step
        const __bf16* ap = hid + (tokg * 16 + l16) * HD_STR + ks * 32 + lhalf * 8;
        bf16x16 a = ld_frag(ap, 16);
        #pragma unroll
        for (int nt = 0; nt < 4; ++nt) {
          const __bf16* bp = w2t + (hq * 64 + nt * 16 + l16) * W2_STR + ks * 32 + lhalf * 8;
          bf16x16 b = ld_frag(bp, 16);
          c2[nt] = wmma_bf16(a, b, c2[nt]);
        }
      }
    }
  }

  // epilogue: contrib[(tok,slot)][h] = gate * (acc + b2[h]); each pair written once
  {
    const int tokg = wid & 1;
    const int hq   = wid >> 1;
    #pragma unroll
    for (int nt = 0; nt < 4; ++nt) {
      const int h = hq * 64 + nt * 16 + l16;
      const float b2v = b2g[e * H_DIM + h];
      #pragma unroll
      for (int r = 0; r < 8; ++r) {
        const int m = tokg * 16 + lhalf * 8 + r;
        const int tok = stok[m];
        if (tok >= 0) {
          contrib[(size_t)(tok * 2 + sslot[m]) * H_DIM + h] = sgate[m] * (c2[nt][r] + b2v);
        }
      }
    }
  }
}

// ------------------------------ combine / loss -----------------------------
__global__ void moe_combine(const float4* __restrict__ contrib4, float4* __restrict__ out4) {
  const int i  = blockIdx.x * 256 + threadIdx.x;  // over T*H/4
  const int t  = i >> 6;                          // token
  const int hq = i & 63;                          // h/4
  float4 a = contrib4[(size_t)t * 128 + hq];
  float4 b = contrib4[(size_t)t * 128 + 64 + hq];
  float4 o;
  o.x = a.x + b.x; o.y = a.y + b.y; o.z = a.z + b.z; o.w = a.w + b.w;
  out4[i] = o;
}

__global__ void moe_lb(const int* __restrict__ counts, float* __restrict__ lb) {
  __shared__ float s[E_NUM];
  const int e = threadIdx.x;
  float u = (float)counts[e] / (float)(T_TOK * 2) - (1.f / (float)E_NUM);
  s[e] = u * u;
  __syncthreads();
  if (e == 0) {
    float t = 0.f;
    for (int i = 0; i < E_NUM; ++i) t += s[i];
    lb[0] = (t / (float)E_NUM) * 0.01f;
  }
}

// ------------------------------ launcher -----------------------------------
extern "C" void kernel_launch(void* const* d_in, const int* in_sizes, int n_in,
                              void* d_out, int out_size, void* d_ws, size_t ws_size,
                              hipStream_t stream) {
  (void)in_sizes; (void)n_in; (void)out_size; (void)ws_size;
  const float* x   = (const float*)d_in[0];
  const float* wr  = (const float*)d_in[1];
  const float* br  = (const float*)d_in[2];
  const float* w1  = (const float*)d_in[3];
  const float* b1  = (const float*)d_in[4];
  const float* w2  = (const float*)d_in[5];
  const float* b2  = (const float*)d_in[6];
  float* out = (float*)d_out;

  char* ws = (char*)d_ws;
  int*   assign_e = (int*)(ws);                                  //  8192 int
  float* assign_g = (float*)(ws + 32768);                        //  8192 f32
  int*   counts   = (int*)(ws + 65536);                          //    64 int
  int*   btok     = (int*)(ws + 65792);                          // 64*4096 int
  float* bgate    = (float*)(ws + 65792 + 1048576);              // 64*4096 f32
  float* contrib  = (float*)(ws + 65792 + 2 * 1048576);          // 2*T*H f32

  hipFuncSetAttribute(reinterpret_cast<const void*>(moe_expert),
                      hipFuncAttributeMaxDynamicSharedMemorySize, SMEM_BYTES);

  moe_router <<<T_TOK, 64, 0, stream>>>(x, wr, br, assign_e, assign_g);
  moe_compact<<<E_NUM, 256, 0, stream>>>(assign_e, assign_g, btok, bgate, counts);
  moe_expert <<<dim3(E_NUM, T_TOK / 32), 256, SMEM_BYTES, stream>>>(
      x, w1, b1, w2, b2, btok, bgate, counts, contrib);
  moe_combine<<<(T_TOK * H_DIM / 4) / 256, 256, 0, stream>>>(
      (const float4*)contrib, (float4*)out);
  moe_lb     <<<1, E_NUM, 0, stream>>>(counts, out + (size_t)T_TOK * H_DIM);
}
